// RefillLoss_3341484557039
// MI455X (gfx1250) — matrix operand order repaired and checked by reading, not compiled, not run on hardware
//
#include <hip/hip_runtime.h>

// Problem constants (match reference)
#define B_  2
#define L_  128
#define LY_ 16
#define E_  256
#define V_  32000
#define C_  18          // 2 static + LY dynamic mixture keys
#define RPB_ 272        // rows per batch in X: L emit + L z + LY y
#define R_   544        // total X rows (B_*RPB_)
#define NT_  2000       // vocab column tiles of 16 (V_/16)

typedef __attribute__((ext_vector_type(16))) __bf16 v16bf;
typedef __attribute__((ext_vector_type(8)))  float  v8f;

__device__ __forceinline__ unsigned short f2bf(float f) {
    unsigned int u = __float_as_uint(f);
    unsigned int r = u + 0x7FFFu + ((u >> 16) & 1u);   // round-to-nearest-even
    return (unsigned short)(r >> 16);
}

// ---------------- Phase 1: precompute / pack ----------------

__global__ void __launch_bounds__(256) cast_embed_kernel(
    const float* __restrict__ emb, unsigned short* __restrict__ Ebf, int n) {
    int i = blockIdx.x * 256 + threadIdx.x;
    if (i < n) Ebf[i] = f2bf(emb[i]);
}

// emit = xsa @ W_emit^T + b_emit ; also write bf16 into X rows [b*272 + l]
__global__ void __launch_bounds__(256) emit_kernel(
    const float* __restrict__ xsa, const float* __restrict__ W_emit,
    const float* __restrict__ b_emit, float* __restrict__ emitf,
    unsigned short* __restrict__ Xbf) {
    __shared__ float sx[E_];
    int r = blockIdx.x;                 // 0..B_*L_-1
    int e = threadIdx.x;
    sx[e] = xsa[r * E_ + e];
    __syncthreads();
    float acc = b_emit[e];
    const float* wrow = W_emit + e * E_;
    #pragma unroll 4
    for (int k = 0; k < E_; ++k) acc = fmaf(sx[k], wrow[k], acc);
    emitf[r * E_ + e] = acc;
    int b = r >> 7, l = r & (L_ - 1);
    Xbf[((size_t)(b * RPB_ + l)) * E_ + e] = f2bf(acc);
}

// xkey rows: c<2 -> W_static rows ; c>=2 -> y @ W_dyn^T + b_dyn
__global__ void __launch_bounds__(256) xkey_kernel(
    const float* __restrict__ y, const float* __restrict__ W_dyn,
    const float* __restrict__ b_dyn, const float* __restrict__ W_static,
    float* __restrict__ xkeyf) {
    __shared__ float sy[E_];
    int bc = blockIdx.x;                // 0..B_*C_-1
    int b = bc / C_, c = bc % C_;
    int e = threadIdx.x;
    if (c < 2) { xkeyf[bc * E_ + e] = W_static[c * E_ + e]; return; }
    sy[e] = y[(b * LY_ + (c - 2)) * E_ + e];
    __syncthreads();
    float acc = b_dyn[e];
    const float* wrow = W_dyn + e * E_;
    #pragma unroll 4
    for (int k = 0; k < E_; ++k) acc = fmaf(sy[k], wrow[k], acc);
    xkeyf[bc * E_ + e] = acc;
}

// pack z and y rows into Xbf (bf16)
__global__ void __launch_bounds__(256) pack_zy_kernel(
    const float* __restrict__ z, const float* __restrict__ y,
    unsigned short* __restrict__ Xbf) {
    int idx = blockIdx.x * 256 + threadIdx.x;
    const int nz = B_ * L_ * E_;        // 65536
    if (idx < nz) {
        int b = idx >> 15, rem = idx & 32767, l = rem >> 8, k = rem & 255;
        Xbf[((size_t)(b * RPB_ + L_ + l)) * E_ + k] = f2bf(z[idx]);
    } else {
        int j = idx - nz;
        if (j < B_ * LY_ * E_) {
            int b = j >> 12, rem = j & 4095, ly = rem >> 8, k = rem & 255;
            Xbf[((size_t)(b * RPB_ + 2 * L_ + ly)) * E_ + k] = f2bf(y[j]);
        }
    }
}

// ---------------- Phase 2: WMMA GEMM + streaming logsumexp ----------------
// A = embed tile (M = 16 vocab rows), B = X rows (N = 16 rows), so the D
// layout gives each lane 8 vocab values of ONE X-row -> per-lane reduction,
// one xor-16 merge. Each wave owns TWO vocab tiles (2 independent WMMA
// accumulation chains sharing each LDS B fragment).
// grid 125 blocks x 256 threads (8 waves); wave w -> vocab tiles 2w, 2w+1.
#define SROWSTRIDE 264

__global__ void __launch_bounds__(256) wmma_logits_kernel(
    const unsigned short* __restrict__ Xbf,   // [R_][E_] bf16 bits
    const unsigned short* __restrict__ Ebf,   // [V_][E_] bf16 bits
    float* __restrict__ pm, float* __restrict__ ps) {
    extern __shared__ __align__(16) unsigned short sX[];   // [272][264]
    const int tid  = threadIdx.x;
    const int lane = tid & 31;
    const int wg   = blockIdx.x * 8 + (tid >> 5);  // 0..999
    const int ct0  = wg * 2;                       // first vocab tile
    const int N    = lane & 15;
    const int half = lane >> 4;

    // A fragments (embed rows, held in VGPRs across both batches):
    // 16-bit A 16x32 layout: lane M = lane&15; VGPR0-3 = K[8*half..+7],
    // VGPR4-7 = K[16+8*half..+7]  -> two contiguous 16B global loads per kb.
    v16bf afrag[2][8];
    #pragma unroll
    for (int t = 0; t < 2; ++t) {
        const unsigned short* erow = Ebf + ((size_t)((ct0 + t) * 16 + N)) * E_;
        #pragma unroll
        for (int kb = 0; kb < 8; ++kb) {
            union { v16bf v; uint4 q[2]; } a;
            a.q[0] = *reinterpret_cast<const uint4*>(erow + kb * 32 + half * 8);
            a.q[1] = *reinterpret_cast<const uint4*>(erow + kb * 32 + 16 + half * 8);
            afrag[t][kb] = a.v;
        }
    }

    for (int b = 0; b < B_; ++b) {
        __syncthreads();
        // async fill of LDS with 272 X rows of this batch (ASYNCcnt path)
        {
            const unsigned short* srcb = Xbf + (size_t)b * RPB_ * E_;
            const uint4* src = reinterpret_cast<const uint4*>(srcb);
            for (int idx = tid; idx < RPB_ * (E_ / 8); idx += 256) {
                int row = idx >> 5, k8 = idx & 31;
                unsigned int ldsaddr =
                    (unsigned int)(size_t)(sX + row * SROWSTRIDE) + k8 * 16;
                const uint4* g = src + row * 32 + k8;
                asm volatile("global_load_async_to_lds_b128 %0, %1, off"
                             :: "v"(ldsaddr), "v"(g) : "memory");
            }
            asm volatile("s_wait_asynccnt 0x0" ::: "memory");
        }
        __syncthreads();

        for (int rt = 0; rt < RPB_ / 16; ++rt) {       // 17 row tiles
            v8f acc0 = {0.f, 0.f, 0.f, 0.f, 0.f, 0.f, 0.f, 0.f};
            v8f acc1 = {0.f, 0.f, 0.f, 0.f, 0.f, 0.f, 0.f, 0.f};
            // 16-bit B 32x16 layout: lane N = lane&15 holds K = kb*32 + 8*half .. +7
            // and kb*32 + 16 + 8*half .. +7 of X row (rt*16+N) -> two 16B LDS loads.
            const unsigned short* brow =
                sX + (rt * 16 + N) * SROWSTRIDE + half * 8;
            #pragma unroll
            for (int kb = 0; kb < 8; ++kb) {
                union { v16bf v; uint4 q[2]; } bv;
                bv.q[0] = *reinterpret_cast<const uint4*>(brow + kb * 32);
                bv.q[1] = *reinterpret_cast<const uint4*>(brow + kb * 32 + 16);
                acc0 = __builtin_amdgcn_wmma_f32_16x16x32_bf16(
                           false, afrag[0][kb], false, bv.v, (short)0, acc0, false, false);
                acc1 = __builtin_amdgcn_wmma_f32_16x16x32_bf16(
                           false, afrag[1][kb], false, bv.v, (short)0, acc1, false, false);
            }
            // per-lane: 8 vocab values of X row (rt*16 + N); reduce in-lane,
            // then merge the two halves (M=0..7 vs M=8..15) with one xor-16.
            #pragma unroll
            for (int t = 0; t < 2; ++t) {
                v8f acc = (t == 0) ? acc0 : acc1;
                float m = acc[0];
                #pragma unroll
                for (int j = 1; j < 8; ++j) m = fmaxf(m, acc[j]);
                float s = 0.f;
                #pragma unroll
                for (int j = 0; j < 8; ++j) s += __expf(acc[j] - m);
                float mo = __shfl_xor(m, 16, 32);
                float so = __shfl_xor(s, 16, 32);
                float M = fmaxf(m, mo);
                float S = s * __expf(m - M) + so * __expf(mo - M);
                if (lane < 16) {
                    int row = b * RPB_ + rt * 16 + lane;
                    pm[(size_t)row * NT_ + (ct0 + t)] = M;
                    ps[(size_t)row * NT_ + (ct0 + t)] = S;
                }
            }
        }
    }
}

// merge 2000 tile partials per row -> logZ
__global__ void __launch_bounds__(256) reduce_logz_kernel(
    const float* __restrict__ pm, const float* __restrict__ ps,
    float* __restrict__ logZ) {
    __shared__ float sm[256], ss[256];
    int r = blockIdx.x, t = threadIdx.x;
    float m = -INFINITY, s = 0.f;
    for (int w = t; w < NT_; w += 256) {
        float m2 = pm[(size_t)r * NT_ + w], s2 = ps[(size_t)r * NT_ + w];
        float M = fmaxf(m, m2), S = 0.f;
        if (m  > -INFINITY) S += s  * __expf(m  - M);
        if (m2 > -INFINITY) S += s2 * __expf(m2 - M);
        m = M; s = S;
    }
    sm[t] = m; ss[t] = s;
    __syncthreads();
    for (int st = 128; st > 0; st >>= 1) {
        if (t < st) {
            float m1 = sm[t], s1 = ss[t], m2 = sm[t + st], s2 = ss[t + st];
            float M = fmaxf(m1, m2), S = 0.f;
            if (m1 > -INFINITY) S += s1 * __expf(m1 - M);
            if (m2 > -INFINITY) S += s2 * __expf(m2 - M);
            sm[t] = M; ss[t] = S;
        }
        __syncthreads();
    }
    if (t == 0) logZ[r] = sm[0] + __logf(ss[0]);
}

// per (b,l): selector log-softmax + target-column logits + mixture logsumexp
__global__ void __launch_bounds__(256) cent_kernel(
    const float* __restrict__ xsa, const float* __restrict__ z,
    const float* __restrict__ y, const long long* __restrict__ yt,
    const float* __restrict__ emb, const float* __restrict__ emitf,
    const float* __restrict__ xkeyf, const float* __restrict__ logZ,
    float* __restrict__ cent) {
    __shared__ float sxsa[E_], sew[E_], selArr[C_], tdArr[C_];
    int bl = blockIdx.x;                    // b*L_ + l
    int b = bl >> 7, l = bl & (L_ - 1);
    int t = threadIdx.x, lane = t & 31, wave = t >> 5;
    sxsa[t] = xsa[(size_t)bl * E_ + t];
    long long tv = yt[bl];
    sew[t] = emb[(size_t)tv * E_ + t];
    __syncthreads();
    for (int c = wave; c < C_; c += 8) {
        const float* key = xkeyf + (size_t)(b * C_ + c) * E_;
        const float* cnd = (c == 0) ? emitf + (size_t)bl * E_
                         : (c == 1) ? z + (size_t)bl * E_
                                    : y + (size_t)(b * LY_ + (c - 2)) * E_;
        float s1 = 0.f, s2 = 0.f;
        for (int k = lane; k < E_; k += 32) {
            s1 = fmaf(sxsa[k], key[k], s1);
            s2 = fmaf(cnd[k], sew[k], s2);
        }
        #pragma unroll
        for (int msk = 16; msk >= 1; msk >>= 1) {
            s1 += __shfl_xor(s1, msk, 32);
            s2 += __shfl_xor(s2, msk, 32);
        }
        if (lane == 0) { selArr[c] = s1; tdArr[c] = s2; }
    }
    __syncthreads();
    if (t == 0) {
        float mx = -INFINITY;
        for (int c = 0; c < C_; ++c) mx = fmaxf(mx, selArr[c]);
        float se = 0.f;
        for (int c = 0; c < C_; ++c) se += __expf(selArr[c] - mx);
        float lse = mx + __logf(se);
        float vals[C_], m2 = -INFINITY;
        for (int c = 0; c < C_; ++c) {
            int rc = (c == 0) ? b * RPB_ + l
                   : (c == 1) ? b * RPB_ + L_ + l
                              : b * RPB_ + 2 * L_ + (c - 2);
            vals[c] = (selArr[c] - lse) + tdArr[c] - logZ[rc];
            m2 = fmaxf(m2, vals[c]);
        }
        float s = 0.f;
        for (int c = 0; c < C_; ++c) s += __expf(vals[c] - m2);
        cent[bl] = m2 + __logf(s);
    }
}

__global__ void __launch_bounds__(128) finalize_kernel(
    const float* __restrict__ cent, float* __restrict__ out) {
    __shared__ float s[128];
    int b = blockIdx.x, t = threadIdx.x;
    s[t] = cent[b * L_ + t];
    __syncthreads();
    for (int st = 64; st > 0; st >>= 1) {
        if (t < st) s[t] += s[t + st];
        __syncthreads();
    }
    if (t == 0) out[b] = -s[0] / (float)L_;
}

// ---------------- Launch ----------------

extern "C" void kernel_launch(void* const* d_in, const int* in_sizes, int n_in,
                              void* d_out, int out_size, void* d_ws, size_t ws_size,
                              hipStream_t stream) {
    const float*     xsa      = (const float*)d_in[0];
    const float*     z        = (const float*)d_in[1];
    const float*     y        = (const float*)d_in[2];
    const long long* yt       = (const long long*)d_in[3];
    const float*     emb      = (const float*)d_in[4];
    const float*     W_static = (const float*)d_in[5];
    const float*     W_dyn    = (const float*)d_in[6];
    const float*     b_dyn    = (const float*)d_in[7];
    const float*     W_emit   = (const float*)d_in[8];
    const float*     b_emit   = (const float*)d_in[9];
    float* out = (float*)d_out;

    char* w = (char*)d_ws;
    size_t off = 0;
    auto take = [&](size_t bytes) -> char* {
        char* p = w + off;
        off += (bytes + 255) & ~(size_t)255;
        return p;
    };
    unsigned short* Ebf   = (unsigned short*)take((size_t)V_ * E_ * 2);
    unsigned short* Xbf   = (unsigned short*)take((size_t)R_ * E_ * 2);
    float*          emitf = (float*)take((size_t)B_ * L_ * E_ * 4);
    float*          xkeyf = (float*)take((size_t)B_ * C_ * E_ * 4);
    float*          pm    = (float*)take((size_t)R_ * NT_ * 4);
    float*          ps    = (float*)take((size_t)R_ * NT_ * 4);
    float*          logZ  = (float*)take((size_t)R_ * 4);
    float*          cent  = (float*)take((size_t)B_ * L_ * 4);

    cast_embed_kernel<<<(V_ * E_) / 256, 256, 0, stream>>>(emb, Ebf, V_ * E_);
    emit_kernel<<<B_ * L_, 256, 0, stream>>>(xsa, W_emit, b_emit, emitf, Xbf);
    xkey_kernel<<<B_ * C_, 256, 0, stream>>>(y, W_dyn, b_dyn, W_static, xkeyf);
    pack_zy_kernel<<<(B_ * L_ * E_ + B_ * LY_ * E_) / 256, 256, 0, stream>>>(z, y, Xbf);

    size_t lds_bytes = (size_t)RPB_ * SROWSTRIDE * 2;   // 143,616 B
    wmma_logits_kernel<<<NT_ / 16, 256, lds_bytes, stream>>>(Xbf, Ebf, pm, ps);

    reduce_logz_kernel<<<R_, 256, 0, stream>>>(pm, ps, logZ);
    cent_kernel<<<B_ * L_, 256, 0, stream>>>(xsa, z, y, yt, emb, emitf, xkeyf, logZ, cent);
    finalize_kernel<<<B_, 128, 0, stream>>>(cent, out);
}
